// CrossAttention_24172075942283
// MI455X (gfx1250) — compile-verified, matrix operand order checked by reading, not run on hardware
//
#include <hip/hip_runtime.h>
#include <hip/hip_bf16.h>

// CDNA5 (gfx1250) cross-attention.
//  k0: weight transpose f32->f16   (Wt[n][k], 3x512x512)
//  k1: Q/K/V projections via v_wmma_f32_16x16x32_f16 (V stored transposed)
//  k2: flash attention, TDM double-buffered K/V tiles in LDS
//
// ws layout: Q f16 [8*2048*512] | K f16 [same] | Vt f16 [B][512][2048] | Wt f16 [3][512][512]

typedef __attribute__((ext_vector_type(16))) _Float16 v16h;
typedef __attribute__((ext_vector_type(8)))  float    v8f;
typedef __attribute__((ext_vector_type(4)))  unsigned int u32x4;
typedef __attribute__((ext_vector_type(8)))  int      i32x8;
typedef __attribute__((ext_vector_type(4)))  int      i32x4;

#define D_MODEL 512
#define SEQ     2048
#define NB      8
#define SOFT_SCALE 0.044194173824159216f   // 1/sqrt(512)

#if __has_builtin(__builtin_amdgcn_tensor_load_to_lds) && \
    __has_builtin(__builtin_amdgcn_s_wait_tensorcnt)
#define HAVE_TDM 1
#else
#define HAVE_TDM 0
#endif

union FragA { v16h h; _Float16 e[16]; uint4 u[2]; };

// element j of a 16-bit A/B fragment covers K = kb + off + kmap(j)
__device__ __forceinline__ int kmap(int j) { return (j & 7) + ((j >> 3) << 4); }

#if HAVE_TDM
// 2-D TDM load of f16 data: tile (tile_d0 contiguous x tile_d1 rows), row
// stride stride0 (elements), into contiguous LDS at lds_off.
// This toolchain's builtin takes 6 args (clang-23 form):
//   (u32x4 g0, i32x8 g1, i32x4 g2, i32x4 g3, i32x8 extra, i32 cpol)
__device__ __forceinline__ void tdm_load_2d_f16(unsigned lds_off, const void* gaddr,
                                                unsigned tensor_d0, unsigned tensor_d1,
                                                unsigned tile_d0, unsigned tile_d1,
                                                unsigned long long stride0)
{
    unsigned long long ga = (unsigned long long)(uintptr_t)gaddr;
    u32x4 g0;
    g0[0] = 1u;                                        // count=1, user descriptor
    g0[1] = lds_off;                                   // LDS byte address
    g0[2] = (unsigned)(ga & 0xffffffffu);              // global_addr[31:0]
    g0[3] = (unsigned)((ga >> 32) & 0x01ffffffu) | (2u << 30);  // [56:32], type=2
    i32x8 g1;
    g1[0] = (int)(1u << 16);                           // data_size=1 (2B), mask=0
    g1[1] = (int)((tensor_d0 & 0xffffu) << 16);        // tensor_dim0[15:0]
    g1[2] = (int)(((tensor_d0 >> 16) & 0xffffu) | ((tensor_d1 & 0xffffu) << 16));
    g1[3] = (int)(((tensor_d1 >> 16) & 0xffffu) | ((tile_d0 & 0xffffu) << 16));
    g1[4] = (int)(tile_d1 & 0xffffu);                  // tile_dim1, tile_dim2=0
    g1[5] = (int)(unsigned)(stride0 & 0xffffffffu);    // tensor_dim0_stride[31:0]
    g1[6] = (int)(unsigned)((stride0 >> 32) & 0xffffu);
    g1[7] = 0;
    i32x4 z4 = {0, 0, 0, 0};
    i32x8 z8 = {0, 0, 0, 0, 0, 0, 0, 0};
    __builtin_amdgcn_tensor_load_to_lds(g0, g1, z4, z4, z8, 0);
}
#endif

// ---------------------------------------------------------------------------
// Kernel 0: Wt[z][n][k] = (f16) W_z[k][n].  grid = (1024, 3), block = 256.
// ---------------------------------------------------------------------------
__global__ __launch_bounds__(256)
void wt_kernel(const float* __restrict__ Wq, const float* __restrict__ Wk,
               const float* __restrict__ Wv, _Float16* __restrict__ Wt)
{
    const int z = blockIdx.y;
    const float* W = (z == 0) ? Wq : (z == 1) ? Wk : Wv;
    _Float16* O = Wt + (size_t)z * D_MODEL * D_MODEL;
    int idx = blockIdx.x * 256 + threadIdx.x;          // [0, 512*512)
    int k = idx >> 9, n = idx & 511;
    O[(size_t)n * D_MODEL + k] = (_Float16)W[idx];     // coalesced read
}

// ---------------------------------------------------------------------------
// Kernel 1: Y = X @ W + b (f16 out).  grid = (1024, 4, 3), block = 256.
// z: 0 -> Q (from x1), 1 -> K (from x2), 2 -> V (from x2, stored transposed)
// ---------------------------------------------------------------------------
__global__ __launch_bounds__(256)
void proj_kernel(const float* __restrict__ x1, const float* __restrict__ x2,
                 const _Float16* __restrict__ Wt,
                 const float* __restrict__ bq, const float* __restrict__ bk,
                 const float* __restrict__ bv,
                 _Float16* __restrict__ Qo, _Float16* __restrict__ Ko,
                 _Float16* __restrict__ Vt)
{
    __shared__ _Float16 Xs[16 * D_MODEL];

    const int z = blockIdx.z;
    const float* __restrict__ X    = (z == 0) ? x1 : x2;
    const float* __restrict__ bias = (z == 0) ? bq : (z == 1) ? bk : bv;

    const int t = threadIdx.x;
    const int rowBase = blockIdx.x * 16;

    const float* xsrc = X + (size_t)rowBase * D_MODEL;
    for (int i = 0; i < 32; ++i) {
        int e = i * 256 + t;
        Xs[e] = (_Float16)xsrc[e];
    }
    __syncthreads();

    const int wave = t >> 5;
    const int lane = t & 31;
    const int hi   = lane >> 4;
    const int off  = hi * 8;
    const int arow = lane & 15;
    const int n    = (blockIdx.y * 8 + wave) * 16 + arow;

    const _Float16* Wtz = Wt + ((size_t)z * D_MODEL + n) * D_MODEL;

    v8f acc = {};
    for (int kc = 0; kc < 16; ++kc) {
        const int kb = kc * 32;
        FragA a, bm;
        a.u[0]  = *(const uint4*)&Xs[arow * D_MODEL + kb + off];
        a.u[1]  = *(const uint4*)&Xs[arow * D_MODEL + kb + off + 16];
        bm.u[0] = *(const uint4*)&Wtz[kb + off];        // Wt row = contiguous K
        bm.u[1] = *(const uint4*)&Wtz[kb + off + 16];
        acc = __builtin_amdgcn_wmma_f32_16x16x32_f16(false, a.h, false, bm.h,
                                                     (short)0, acc, false, false);
    }

    const float bn = bias[n];
    #pragma unroll
    for (int r = 0; r < 8; ++r) {
        int row = rowBase + r + hi * 8;                 // global row = b*2048 + pos
        float v = acc[r] + bn;
        if (z == 2) {                                   // V stored transposed
            int batch = row >> 11, key = row & 2047;
            Vt[((size_t)batch * D_MODEL + n) * SEQ + key] = (_Float16)v;
        } else {
            _Float16* O = (z == 0) ? Qo : Ko;
            O[(size_t)row * D_MODEL + n] = (_Float16)v;
        }
    }
}

// ---------------------------------------------------------------------------
// Kernel 2: flash attention.  grid = (128, 8), block = 128 (4 waves).
// TDM double-buffers the 32-key K tile (32x512) and V^T tile (512x32) in LDS.
// ---------------------------------------------------------------------------
__global__ __launch_bounds__(128)
void attn_kernel(const _Float16* __restrict__ Qm, const _Float16* __restrict__ Km,
                 const _Float16* __restrict__ Vt, float* __restrict__ out)
{
    __shared__ _Float16 Qs[16 * D_MODEL];        // 16 KB
    __shared__ _Float16 Ks[2][32 * D_MODEL];     // 2 x 32 KB
    __shared__ _Float16 Vts[2][D_MODEL * 32];    // 2 x 32 KB  (col-major V tile)
    __shared__ float    Sls[16 * 32];
    __shared__ _Float16 Ps[16 * 32];
    __shared__ float    rowM[16], rowL[16], rowScale[16];

    const int t    = threadIdx.x;
    const int wave = t >> 5;
    const int lane = t & 31;
    const int hi   = lane >> 4;
    const int off  = hi * 8;
    const int arow = lane & 15;

    const int bb = blockIdx.y;
    const int q0 = blockIdx.x * 16;
    const _Float16* Kb  = Km + (size_t)bb * SEQ * D_MODEL;
    const _Float16* Vtb = Vt + (size_t)bb * D_MODEL * SEQ;
    const int NT = SEQ / 32;

#if HAVE_TDM
    if (wave == 0) {        // prime buffer 0 (TDM ignores EXEC; once per wave)
        tdm_load_2d_f16((unsigned)(uintptr_t)&Ks[0][0],  Kb,
                        D_MODEL, SEQ, D_MODEL, 32, D_MODEL);
        tdm_load_2d_f16((unsigned)(uintptr_t)&Vts[0][0], Vtb,
                        SEQ, D_MODEL, 32, D_MODEL, SEQ);
    }
#endif

    {   // Q tile: 8192 halfs = 1024 uint4
        const uint4* src = (const uint4*)(Qm + ((size_t)bb * SEQ + q0) * D_MODEL);
        uint4* dst = (uint4*)Qs;
        for (int i = 0; i < 8; ++i) dst[i * 128 + t] = src[i * 128 + t];
    }
    if (t < 16) { rowM[t] = -1e30f; rowL[t] = 0.f; }

    v8f o[8] = {};

    for (int kt = 0; kt < NT; ++kt) {
        const int buf = kt & 1;
#if HAVE_TDM
        if (wave == 0) __builtin_amdgcn_s_wait_tensorcnt(0);  // tile kt resident
        __syncthreads();   // data visible to all waves; prev reads of buf^1 done
        if (wave == 0 && kt + 1 < NT) {   // overlap DMA of kt+1 with compute of kt
            const size_t koff = (size_t)(kt + 1) * 32;
            tdm_load_2d_f16((unsigned)(uintptr_t)&Ks[buf ^ 1][0],  Kb + koff * D_MODEL,
                            D_MODEL, SEQ, D_MODEL, 32, D_MODEL);
            tdm_load_2d_f16((unsigned)(uintptr_t)&Vts[buf ^ 1][0], Vtb + koff,
                            SEQ, D_MODEL, 32, D_MODEL, SEQ);
        }
#else
        __syncthreads();
        {   // manual stage of tile kt
            const uint4* ksrc = (const uint4*)(Kb + (size_t)kt * 32 * D_MODEL);
            uint4* kdst = (uint4*)&Ks[buf][0];
            for (int i = 0; i < 16; ++i) kdst[i * 128 + t] = ksrc[i * 128 + t];
            const uint4* vsrc = (const uint4*)Vtb;       // col stride = 256 uint4
            uint4* vdst = (uint4*)&Vts[buf][0];
            for (int i = 0; i < 16; ++i) {
                int ii = i * 128 + t, n = ii >> 2, seg = ii & 3;
                vdst[ii] = vsrc[(size_t)n * (SEQ / 8) + kt * 4 + seg];
            }
        }
#endif
        for (int i = 0; i < 4; ++i) Sls[i * 128 + t] = 0.f;
        __syncthreads();

        // ---- partial S = Q K^T; wave w covers d-chunks [4w, 4w+4) ----
        v8f s0 = {}, s1 = {};
        #pragma unroll
        for (int kc = 0; kc < 4; ++kc) {
            const int kb = (wave * 4 + kc) * 32;
            FragA a, b0, b1;
            a.u[0]  = *(const uint4*)&Qs[arow * D_MODEL + kb + off];
            a.u[1]  = *(const uint4*)&Qs[arow * D_MODEL + kb + off + 16];
            b0.u[0] = *(const uint4*)&Ks[buf][arow * D_MODEL + kb + off];
            b0.u[1] = *(const uint4*)&Ks[buf][arow * D_MODEL + kb + off + 16];
            b1.u[0] = *(const uint4*)&Ks[buf][(16 + arow) * D_MODEL + kb + off];
            b1.u[1] = *(const uint4*)&Ks[buf][(16 + arow) * D_MODEL + kb + off + 16];
            s0 = __builtin_amdgcn_wmma_f32_16x16x32_f16(false, a.h, false, b0.h,
                                                        (short)0, s0, false, false);
            s1 = __builtin_amdgcn_wmma_f32_16x16x32_f16(false, a.h, false, b1.h,
                                                        (short)0, s1, false, false);
        }
        #pragma unroll
        for (int r = 0; r < 8; ++r) {
            int m = r + hi * 8;
            atomicAdd(&Sls[m * 32 + arow],      s0[r]);   // ds_add_f32
            atomicAdd(&Sls[m * 32 + 16 + arow], s1[r]);
        }
        __syncthreads();

        // ---- online softmax, one lane per query row ----
        if (t < 16) {
            float mOld = rowM[t], mNew = mOld;
            float sv[32];
            #pragma unroll
            for (int j = 0; j < 32; ++j) {
                sv[j] = Sls[t * 32 + j] * SOFT_SCALE;
                mNew  = fmaxf(mNew, sv[j]);
            }
            float sum = 0.f;
            #pragma unroll
            for (int j = 0; j < 32; ++j) {
                float p = __expf(sv[j] - mNew);
                Ps[t * 32 + j] = (_Float16)p;
                sum += p;
            }
            float sc = __expf(mOld - mNew);
            rowScale[t] = sc;
            rowL[t] = rowL[t] * sc + sum;
            rowM[t] = mNew;
        }
        __syncthreads();

        // ---- O = O*scale + P @ V ----
        float sc[8];
        #pragma unroll
        for (int r = 0; r < 8; ++r) sc[r] = rowScale[r + hi * 8];

        FragA pa;                       // P in A-layout via LDS
        pa.u[0] = *(const uint4*)&Ps[arow * 32 + off];
        pa.u[1] = *(const uint4*)&Ps[arow * 32 + off + 16];

        #pragma unroll
        for (int f = 0; f < 8; ++f) {
            #pragma unroll
            for (int r = 0; r < 8; ++r) o[f][r] *= sc[r];
            const int n = wave * 128 + f * 16 + arow;
            FragA vb;                   // V^T tile: contiguous keys per column
            vb.u[0] = *(const uint4*)&Vts[buf][n * 32 + off];
            vb.u[1] = *(const uint4*)&Vts[buf][n * 32 + off + 16];
            o[f] = __builtin_amdgcn_wmma_f32_16x16x32_f16(false, pa.h, false, vb.h,
                                                          (short)0, o[f], false, false);
        }
    }

    __syncthreads();
    float inv[8];
    #pragma unroll
    for (int r = 0; r < 8; ++r) inv[r] = 1.f / rowL[r + hi * 8];
    #pragma unroll
    for (int f = 0; f < 8; ++f) {
        const int n = wave * 128 + f * 16 + arow;
        #pragma unroll
        for (int r = 0; r < 8; ++r) {
            int row = q0 + r + hi * 8;
            out[((size_t)bb * SEQ + row) * D_MODEL + n] = o[f][r] * inv[r];
        }
    }
}

// ---------------------------------------------------------------------------
extern "C" void kernel_launch(void* const* d_in, const int* in_sizes, int n_in,
                              void* d_out, int out_size, void* d_ws, size_t ws_size,
                              hipStream_t stream)
{
    (void)in_sizes; (void)n_in; (void)out_size; (void)ws_size;
    const float* x1 = (const float*)d_in[0];
    const float* x2 = (const float*)d_in[1];
    const float* Wq = (const float*)d_in[2];
    const float* bq = (const float*)d_in[3];
    const float* Wk = (const float*)d_in[4];
    const float* bk = (const float*)d_in[5];
    const float* Wv = (const float*)d_in[6];
    const float* bv = (const float*)d_in[7];
    float* out = (float*)d_out;

    const size_t mat = (size_t)NB * SEQ * D_MODEL;
    _Float16* Qh = (_Float16*)d_ws;
    _Float16* Kh = Qh + mat;
    _Float16* Vt = Kh + mat;
    _Float16* Wt = Vt + mat;            // 3*512*512 f16

    wt_kernel<<<dim3(D_MODEL * D_MODEL / 256, 3), 256, 0, stream>>>(Wq, Wk, Wv, Wt);

    dim3 g1(NB * SEQ / 16, 4, 3);
    proj_kernel<<<g1, 256, 0, stream>>>(x1, x2, Wt, bq, bk, bv, Qh, Kh, Vt);

    dim3 g2(SEQ / 16, NB);
    attn_kernel<<<g2, 128, 0, stream>>>(Qh, Kh, Vt, out);
}
